// GraphSAGEEncoder_22703197126666
// MI455X (gfx1250) — compile-verified, matrix operand order checked by reading, not run on hardware
//
#include <hip/hip_runtime.h>
#include <hip/hip_bf16.h>
#include <math.h>

// GraphSAGE encoder for MI455X (gfx1250): WMMA f16->f32 GEMMs + atomic
// scatter-logsumexp + fused LN/ReLU/residual.

#define D_H     64
#define TAU     1.0f
#define SEG_EPS 1e-30f
#define LN_EPS  1e-5f
#define ALPHA   0.5f

typedef __attribute__((ext_vector_type(16))) _Float16 v16h;
typedef __attribute__((ext_vector_type(8)))  float    v8f;

// ---------------------------------------------------------------------------
// float atomic max via signed-max / unsigned-min bit ordering (portable,
// lowers to global_atomic_max_i32 / global_atomic_min_u32)
// ---------------------------------------------------------------------------
__device__ __forceinline__ void atomic_max_float(float* addr, float value) {
    int iv = __float_as_int(value);
    if (iv >= 0) atomicMax((int*)addr, iv);
    else         atomicMin((unsigned int*)addr, (unsigned int)iv);
}

// ---------------------------------------------------------------------------
// WMMA GEMM:  out[r, j] = sum_k A[r,k] * W[j,k] + bias[j]   (out is nrows x 64)
// A = concat(A0, A1) along k when SPLIT; both have row stride 64.
// One wave -> one 16x16 output tile. blockDim = 128 (4 waves = 64 out cols).
// K compile-time (64 or 128) so the loop fully unrolls into 2/4 WMMA ops.
// ---------------------------------------------------------------------------
template <int K, bool SPLIT, bool GUARD>
__global__ void wmma_gemm_kernel(const float* __restrict__ A0,
                                 const float* __restrict__ A1,
                                 const float* __restrict__ W,
                                 const float* __restrict__ bias,
                                 float* __restrict__ out,
                                 int nrows) {
    constexpr int K0  = 64;            // k-extent of A0 (= row stride of A0/A1)
    const int lane    = threadIdx.x & 31;
    const int wave    = threadIdx.x >> 5;
    const int rowBase = blockIdx.x << 4;
    const int colBase = wave << 4;
    const int mr      = lane & 15;     // A-row / B-col within tile
    const int hi      = lane >> 4;     // K sub-chunk selector

    int r = rowBase + mr;
    if (GUARD && r >= nrows) r = nrows - 1;  // clamp loads; EXEC stays all-1

    v8f c = {};
    #pragma unroll
    for (int ks = 0; ks < K; ks += 32) {
        const bool first = (!SPLIT) || (ks < K0);
        const float* arow = first ? (A0 + (size_t)r * K0) : (A1 + (size_t)r * K0);
        const int kloc = first ? ks : (ks - K0);
        // A lane layout (ISA 7.12.2, 16-bit A 16x32): halves[0..7] = K = hi*8+j,
        // halves[8..15] = K = 16 + hi*8 + j  (relative to chunk base ks)
        const float* pa = arow + kloc + (hi << 3);
        // B mirrors A with column index; W is row-major [64][K], B[k][j] = W[j][k]
        const float* pb = W + (size_t)(colBase + mr) * K + ks + (hi << 3);
        v16h a, b;
        #pragma unroll
        for (int j = 0; j < 8; ++j) {
            a[j]     = (_Float16)pa[j];
            a[8 + j] = (_Float16)pa[16 + j];
            b[j]     = (_Float16)pb[j];
            b[8 + j] = (_Float16)pb[16 + j];
        }
        c = __builtin_amdgcn_wmma_f32_16x16x32_f16(false, a, false, b,
                                                   (short)0, c, false, false);
    }

    // C/D layout: VGPR v holds M = v + 8*hi, N = lane%16
    const int   ocol = colBase + mr;
    const float bv   = bias[ocol];
    #pragma unroll
    for (int v = 0; v < 8; ++v) {
        const int orow = rowBase + v + (hi << 3);
        if (!GUARD || orow < nrows)
            out[(size_t)orow * D_H + ocol] = c[v] + bv;
    }
}

// ---------------------------------------------------------------------------
// Scatter logsumexp, pass 0: init buffers
// ---------------------------------------------------------------------------
__global__ void init_bufs_kernel(float* __restrict__ maxbuf,
                                 float* __restrict__ sumbuf,
                                 int*   __restrict__ deg,
                                 int n64, int n) {
    int i = blockIdx.x * blockDim.x + threadIdx.x;
    if (i < n64) {
        maxbuf[i] = -INFINITY;
        sumbuf[i] = 0.0f;
    }
    if (i < n) deg[i] = 0;
}

// pass 1: per-(edge, 4-dim chunk): vector load h[src], atomic max into dst row
__global__ void scatter_max_kernel(const float* __restrict__ h,
                                   const int* __restrict__ esrc,
                                   const int* __restrict__ edst,
                                   float* __restrict__ maxbuf,
                                   int*   __restrict__ deg,
                                   long long total16) {
    long long i = (long long)blockIdx.x * blockDim.x + threadIdx.x;
    if (i >= total16) return;
    const int e  = (int)(i >> 4);
    const int d4 = (int)(i & 15) << 2;
    const int s = esrc[e];
    const int t = edst[e];
    const float4 v = *(const float4*)(h + (size_t)s * D_H + d4);
    float* mrow = maxbuf + (size_t)t * D_H + d4;
    atomic_max_float(mrow + 0, v.x * (1.0f / TAU));
    atomic_max_float(mrow + 1, v.y * (1.0f / TAU));
    atomic_max_float(mrow + 2, v.z * (1.0f / TAU));
    atomic_max_float(mrow + 3, v.w * (1.0f / TAU));
    if (d4 == 0) atomicAdd(&deg[t], 1);
}

// pass 2: atomic sum of exp(h[src]/tau - max[dst]), 4 dims per thread
__global__ void scatter_sumexp_kernel(const float* __restrict__ h,
                                      const int* __restrict__ esrc,
                                      const int* __restrict__ edst,
                                      const float* __restrict__ maxbuf,
                                      float* __restrict__ sumbuf,
                                      long long total16) {
    long long i = (long long)blockIdx.x * blockDim.x + threadIdx.x;
    if (i >= total16) return;
    const int e  = (int)(i >> 4);
    const int d4 = (int)(i & 15) << 2;
    const int s = esrc[e];
    const int t = edst[e];
    const float4 v = *(const float4*)(h + (size_t)s * D_H + d4);
    const float4 m = *(const float4*)(maxbuf + (size_t)t * D_H + d4);
    float* srow = sumbuf + (size_t)t * D_H + d4;
    atomicAdd(srow + 0, __expf(v.x * (1.0f / TAU) - m.x));
    atomicAdd(srow + 1, __expf(v.y * (1.0f / TAU) - m.y));
    atomicAdd(srow + 2, __expf(v.z * (1.0f / TAU) - m.z));
    atomicAdd(srow + 3, __expf(v.w * (1.0f / TAU) - m.w));
}

// pass 3: agg = deg>0 ? tau*(max + log(max(sum, eps))) : 0   (in-place: sumbuf)
__global__ void finalize_agg_kernel(const float* __restrict__ maxbuf,
                                    const int* __restrict__ deg,
                                    float* __restrict__ sum_to_agg,
                                    int n64) {
    int i = blockIdx.x * blockDim.x + threadIdx.x;
    if (i >= n64) return;
    const int node = i >> 6;
    if (deg[node] > 0) {
        const float m = maxbuf[i];
        const float s = fmaxf(sum_to_agg[i], SEG_EPS);
        sum_to_agg[i] = TAU * (m + __logf(s));
    } else {
        sum_to_agg[i] = 0.0f;
    }
}

// ---------------------------------------------------------------------------
// LayerNorm + ReLU + residual. One wave32 per node, 2 elems/lane.
// out[n] = ALPHA*h_old[n] + (1-ALPHA)*relu(LN(h_pre[n]))
// ---------------------------------------------------------------------------
__global__ void ln_relu_res_kernel(const float* __restrict__ h_old,
                                   const float* __restrict__ h_pre,
                                   const float* __restrict__ gamma,
                                   const float* __restrict__ beta,
                                   float* __restrict__ out,
                                   int n) {
    const int node = blockIdx.x * (blockDim.x >> 5) + (threadIdx.x >> 5);
    if (node >= n) return;
    const int lane = threadIdx.x & 31;
    const float* row = h_pre + (size_t)node * D_H;
    float x0 = row[lane];
    float x1 = row[lane + 32];
    float s  = x0 + x1;
    float sq = x0 * x0 + x1 * x1;
    #pragma unroll
    for (int off = 16; off > 0; off >>= 1) {
        s  += __shfl_xor(s,  off, 32);
        sq += __shfl_xor(sq, off, 32);
    }
    const float mu  = s * (1.0f / D_H);
    const float var = sq * (1.0f / D_H) - mu * mu;
    const float inv = rsqrtf(var + LN_EPS);
    float y0 = (x0 - mu) * inv * gamma[lane]      + beta[lane];
    float y1 = (x1 - mu) * inv * gamma[lane + 32] + beta[lane + 32];
    y0 = fmaxf(y0, 0.0f);
    y1 = fmaxf(y1, 0.0f);
    const float* ho = h_old + (size_t)node * D_H;
    float* po = out + (size_t)node * D_H;
    po[lane]      = ALPHA * ho[lane]      + (1.0f - ALPHA) * y0;
    po[lane + 32] = ALPHA * ho[lane + 32] + (1.0f - ALPHA) * y1;
}

// ---------------------------------------------------------------------------
extern "C" void kernel_launch(void* const* d_in, const int* in_sizes, int n_in,
                              void* d_out, int out_size, void* d_ws, size_t ws_size,
                              hipStream_t stream) {
    const float* x       = (const float*)d_in[0];
    const int*   esrc    = (const int*)  d_in[1];
    const int*   edst    = (const int*)  d_in[2];
    const float* W_in    = (const float*)d_in[3];
    const float* b_in    = (const float*)d_in[4];
    const float* W_l     = (const float*)d_in[5];
    const float* b_l     = (const float*)d_in[6];
    const float* gamma_l = (const float*)d_in[7];
    const float* beta_l  = (const float*)d_in[8];
    float*       out     = (float*)d_out;

    const int N = in_sizes[0] / D_H;     // 50000
    const int E = in_sizes[1];           // 1200000
    const int L = in_sizes[6] / D_H;     // 3
    const int n64 = N * D_H;
    const bool exact = (N % 16) == 0;    // true for N=50000

    // workspace carve-up (~38.6 MB): h | maxbuf (reused as h_new) | sumbuf (->agg) | deg
    char* ws = (char*)d_ws;
    float* h      = (float*)ws; ws += (size_t)n64 * sizeof(float);
    float* maxbuf = (float*)ws; ws += (size_t)n64 * sizeof(float);  // also h_new
    float* sumbuf = (float*)ws; ws += (size_t)n64 * sizeof(float);  // also agg
    int*   deg    = (int*)  ws; ws += (size_t)N   * sizeof(int);
    float* h_new  = maxbuf;  // lifetimes do not overlap within a layer
    float* agg    = sumbuf;

    const int gemmGrid = (N + 15) / 16;       // 3125 row tiles, 4 waves/block
    const long long totalE16 = (long long)E * 16;      // 4 dims per thread
    const int scatterGrid = (int)((totalE16 + 255) / 256);
    const int initGrid = (n64 + 255) / 256;
    const int lnGrid = (N + 7) / 8;           // 8 nodes per 256-thread block

    // h = x @ W_in.T + b_in
    if (exact)
        wmma_gemm_kernel<64, false, false><<<gemmGrid, 128, 0, stream>>>(
            x, nullptr, W_in, b_in, h, N);
    else
        wmma_gemm_kernel<64, false, true><<<gemmGrid, 128, 0, stream>>>(
            x, nullptr, W_in, b_in, h, N);

    for (int l = 0; l < L; ++l) {
        init_bufs_kernel<<<initGrid, 256, 0, stream>>>(maxbuf, sumbuf, deg, n64, N);
        scatter_max_kernel<<<scatterGrid, 256, 0, stream>>>(h, esrc, edst,
                                                            maxbuf, deg, totalE16);
        scatter_sumexp_kernel<<<scatterGrid, 256, 0, stream>>>(h, esrc, edst,
                                                               maxbuf, sumbuf, totalE16);
        finalize_agg_kernel<<<initGrid, 256, 0, stream>>>(maxbuf, deg, agg, n64);

        // h_new = [h ‖ agg] @ W_l[l].T + b_l[l]   (K = 128, split at 64)
        const float* Wl = W_l + (size_t)l * D_H * 2 * D_H;
        const float* bl = b_l + (size_t)l * D_H;
        if (exact)
            wmma_gemm_kernel<128, true, false><<<gemmGrid, 128, 0, stream>>>(
                h, agg, Wl, bl, h_new, N);
        else
            wmma_gemm_kernel<128, true, true><<<gemmGrid, 128, 0, stream>>>(
                h, agg, Wl, bl, h_new, N);

        float* dst = (l == L - 1) ? out : h;
        ln_relu_res_kernel<<<lnGrid, 256, 0, stream>>>(
            h, h_new, gamma_l + (size_t)l * D_H, beta_l + (size_t)l * D_H, dst, N);
    }
}
